// RnnSenderGS_24730421690697
// MI455X (gfx1250) — compile-verified, matrix operand order checked by reading, not run on hardware
//
#include <hip/hip_runtime.h>
#include <hip/hip_bf16.h>

// ---------------------------------------------------------------------------
// RnnSenderGS on MI455X (gfx1250): batch-tiled persistent GRU with WMMA f16.
// B=2048, D_IN=H=E=512, V=1024, L=32. Each workgroup owns 16 batch rows and
// runs all 32 steps. All matmuls via v_wmma_f32_16x16x32_f16; gumbel tiles
// streamed via global_load_async_to_lds_b128 (ASYNCcnt) overlapped with the
// gate GEMMs; outputs stored non-temporal. LDS = 129KB -> 2 blocks/WGP.
// ---------------------------------------------------------------------------

#define B_    2048
#define DIN   512
#define H_    512
#define E_    512
#define V_    1024
#define L_    32
#define H3    1536
#define TILE_B 16
#define THREADS 256
#define INV_TEMP 1.0f

// LDS row strides (elements), padded to avoid bank conflicts on row gathers
#define EH_STRIDE 520   // f16 rows of 512
#define LG_STRIDE 1028  // f32 rows of 1024
#define SB_STRIDE 1032  // f16 rows of 1024

typedef _Float16 half_t;
typedef __attribute__((ext_vector_type(16))) _Float16 v16h;
typedef __attribute__((ext_vector_type(8)))  _Float16 v8h;
typedef __attribute__((ext_vector_type(8)))  float    v8f;

// Load a 16x32 f16 A-fragment (or mirrored B-fragment) for one wave.
// Row index within the 16-row tile = lane&15; lane>=16 takes K-halves 8..15/24..31.
__device__ __forceinline__ v16h load_frag(const half_t* __restrict__ base,
                                          int stride, int lane, int k0) {
  const int hs = lane >> 4;        // 0 or 1
  const int r  = lane & 15;        // tile row (A) / tile col (B)
  const half_t* p = base + r * stride + k0 + hs * 8;
  v8h lo = *(const v8h*)(p);        // K +0..7   (or +8..15 for hs=1)
  v8h hi = *(const v8h*)(p + 16);   // K +16..23 (or +24..31)
  v16h a;
#pragma unroll
  for (int i = 0; i < 8; i++) { a[i] = lo[i]; a[8 + i] = hi[i]; }
  return a;
}

__device__ __forceinline__ v8f wmma16(v16h a, v16h b, v8f c) {
  return __builtin_amdgcn_wmma_f32_16x16x32_f16(
      /*neg_a=*/false, a, /*neg_b=*/false, b,
      /*c_mod=*/(short)0, c, /*reuse_a=*/false, /*reuse_b=*/false);
}

// ----------------------- weight conversion pre-pass ------------------------
__global__ void cvt_copy_f16(const float* __restrict__ src,
                             half_t* __restrict__ dst, int n) {
  int i = blockIdx.x * blockDim.x + threadIdx.x;
  if (i < n) dst[i] = (half_t)src[i];
}

// src [rows][cols] (row-major) -> dst [cols][rows]
__global__ void cvt_transpose_f16(const float* __restrict__ src,
                                  half_t* __restrict__ dst, int rows, int cols) {
  int i = blockIdx.x * blockDim.x + threadIdx.x;
  if (i < rows * cols) {
    int r = i / cols, c = i % cols;
    dst[c * rows + r] = (half_t)src[i];
  }
}

// ------------------------------ main kernel --------------------------------
__global__ __launch_bounds__(THREADS) void rnn_sender_kernel(
    const float* __restrict__ x,     const float* __restrict__ b_agent,
    const float* __restrict__ b_ih,  const float* __restrict__ b_hh,
    const float* __restrict__ b_out, const float* __restrict__ b_emb,
    const float* __restrict__ sos,   const float* __restrict__ gumbel,
    const half_t* __restrict__ Wt_agent,  // [H][DIN]
    const half_t* __restrict__ Wb_ih,     // [3H][E]
    const half_t* __restrict__ Wb_hh,     // [3H][H]
    const half_t* __restrict__ Wt_out,    // [V][H]
    const half_t* __restrict__ Wt_emb,    // [E][V]
    float* __restrict__ out)              // [B][L+1][V]
{
  extern __shared__ char smem_raw[];
  half_t* e_bf = (half_t*)smem_raw;                           // [16][EH_STRIDE]
  half_t* h_bf = e_bf + TILE_B * EH_STRIDE;                   // [16][EH_STRIDE]
  float*  lg   = (float*)(h_bf + TILE_B * EH_STRIDE);         // [16][LG_STRIDE]
  half_t* sm   = (half_t*)(lg + TILE_B * LG_STRIDE);          // [16][SB_STRIDE]

  const int tid   = threadIdx.x;
  const int lane  = tid & 31;
  const int wave  = tid >> 5;
  const int bbase = blockIdx.x * TILE_B;
  const int ncol  = lane & 15;          // fixed N for C/D fragments
  const int mrow  = (lane >> 4) * 8;    // M base for C/D fragments

  // Hardware LDS byte offset of the logits buffer (for async-to-LDS ops).
  const unsigned lgbase =
      (unsigned)__builtin_amdgcn_groupstaticsize() +
      (unsigned)((const char*)lg - (const char*)smem_raw);

  // ---- stage x tile as f16 into sm, then h0 = x @ W_agent(+b) -----------
  for (int i = tid; i < TILE_B * DIN; i += THREADS) {
    int r = i >> 9, c = i & 511;
    sm[r * SB_STRIDE + c] = (half_t)x[(size_t)(bbase + r) * DIN + c];
  }
  __syncthreads();

#pragma unroll
  for (int q = 0; q < 4; q++) {
    const int nb = (wave * 4 + q) * 16;
    v8f acc = {};
    for (int k0 = 0; k0 < DIN; k0 += 32) {
      v16h a = load_frag(sm, SB_STRIDE, lane, k0);
      v16h b = load_frag(Wt_agent + (size_t)nb * DIN, DIN, lane, k0);
      acc = wmma16(a, b, acc);
    }
    const int n = nb + ncol;
    const float bias = b_agent[n];
#pragma unroll
    for (int i = 0; i < 8; i++) {
      int m = mrow + i;
      h_bf[m * EH_STRIDE + n] = (half_t)(acc[i] + bias);
    }
  }
  // e0 = sos broadcast
  for (int i = tid; i < TILE_B * E_; i += THREADS) {
    int r = i >> 9, c = i & 511;
    e_bf[r * EH_STRIDE + c] = (half_t)sos[c];
  }
  __syncthreads();

  // ------------------------------ time loop --------------------------------
  for (int t = 0; t < L_; t++) {
    // ---- async-stream this step's gumbel tile [16][1024] f32 into lg -------
    // Overlaps with the gate GEMMs; drained with s_wait_asynccnt before the
    // barrier that precedes the logits phase.
    {
      const float* gsrc = gumbel + ((size_t)t * B_ + bbase) * V_;
#pragma unroll
      for (int k = 0; k < 16; k++) {
        int idx = tid + k * THREADS;         // 0..4095 (16 rows x 256 x b128)
        int row = idx >> 8;
        int c4  = (idx & 255) * 4;           // float column
        unsigned dst = lgbase + (unsigned)(row * LG_STRIDE + c4) * 4u;
        unsigned long long src =
            (unsigned long long)(const void*)(gsrc + (size_t)row * V_ + c4);
        asm volatile("global_load_async_to_lds_b128 %0, %1, off"
                     :: "v"(dst), "v"(src) : "memory");
      }
    }

    // ---- fused gate GEMMs: r,z need (e·Wih + h·Whh); n needs xn, hn apart --
    v8f hnew_store[4];
#pragma unroll
    for (int q = 0; q < 4; q++) {
      const int jb = (wave * 4 + q) * 16;
      v8f accR = {}, accZ = {}, accXN = {}, accHN = {};
      for (int k0 = 0; k0 < E_; k0 += 32) {
        v16h ae = load_frag(e_bf, EH_STRIDE, lane, k0);
        v16h ah = load_frag(h_bf, EH_STRIDE, lane, k0);
        v16h b0 = load_frag(Wb_ih + (size_t)jb * E_, E_, lane, k0);
        accR  = wmma16(ae, b0, accR);
        v16h b1 = load_frag(Wb_hh + (size_t)jb * H_, H_, lane, k0);
        accR  = wmma16(ah, b1, accR);
        v16h b2 = load_frag(Wb_ih + (size_t)(jb + 512) * E_, E_, lane, k0);
        accZ  = wmma16(ae, b2, accZ);
        v16h b3 = load_frag(Wb_hh + (size_t)(jb + 512) * H_, H_, lane, k0);
        accZ  = wmma16(ah, b3, accZ);
        v16h b4 = load_frag(Wb_ih + (size_t)(jb + 1024) * E_, E_, lane, k0);
        accXN = wmma16(ae, b4, accXN);
        v16h b5 = load_frag(Wb_hh + (size_t)(jb + 1024) * H_, H_, lane, k0);
        accHN = wmma16(ah, b5, accHN);
      }
      const int n   = jb + ncol;
      const float br  = b_ih[n]        + b_hh[n];
      const float bz  = b_ih[n + 512]  + b_hh[n + 512];
      const float bxn = b_ih[n + 1024];
      const float bhn = b_hh[n + 1024];
      v8f hn;
#pragma unroll
      for (int i = 0; i < 8; i++) {
        int m = mrow + i;
        float r  = 1.0f / (1.0f + __expf(-(accR[i] + br)));
        float z  = 1.0f / (1.0f + __expf(-(accZ[i] + bz)));
        float nn = tanhf(accXN[i] + bxn + r * (accHN[i] + bhn));
        float ho = (float)h_bf[m * EH_STRIDE + n];
        hn[i] = (1.0f - z) * nn + z * ho;
      }
      hnew_store[q] = hn;
    }
    __syncthreads();   // all reads of old h done
#pragma unroll
    for (int q = 0; q < 4; q++) {
      const int n = (wave * 4 + q) * 16 + ncol;
#pragma unroll
      for (int i = 0; i < 8; i++) {
        int m = mrow + i;
        h_bf[m * EH_STRIDE + n] = (half_t)hnew_store[q][i];
      }
    }
    // drain this wave's async gumbel copies, then make them WG-visible
    asm volatile("s_wait_asynccnt 0" ::: "memory");
    __syncthreads();

    // ---- logits = h_new @ W_out + b_out + gumbel[t] (gumbel already in lg) --
#pragma unroll
    for (int q = 0; q < 8; q++) {
      const int nb = (wave * 8 + q) * 16;
      v8f acc = {};
      for (int k0 = 0; k0 < H_; k0 += 32) {
        v16h a = load_frag(h_bf, EH_STRIDE, lane, k0);
        v16h b = load_frag(Wt_out + (size_t)nb * H_, H_, lane, k0);
        acc = wmma16(a, b, acc);
      }
      const int v = nb + ncol;
      const float bias = b_out[v];
#pragma unroll
      for (int i = 0; i < 8; i++) {
        int m = mrow + i;
        float g = lg[m * LG_STRIDE + v];
        lg[m * LG_STRIDE + v] = acc[i] + bias + g;
      }
    }
    __syncthreads();

    // ---- softmax per row (2 rows per wave), emit probs + f16 sample ---------
#pragma unroll
    for (int rq = 0; rq < 2; rq++) {
      const int row = wave * 2 + rq;
      const float* lrow = lg + row * LG_STRIDE + lane * 32;
      float vals[32];
      float mx = -3.0e38f;
#pragma unroll
      for (int i = 0; i < 32; i++) { vals[i] = lrow[i]; mx = fmaxf(mx, vals[i]); }
      for (int off = 16; off > 0; off >>= 1) mx = fmaxf(mx, __shfl_xor(mx, off, 32));
      float s = 0.0f;
#pragma unroll
      for (int i = 0; i < 32; i++) { vals[i] = __expf((vals[i] - mx) * INV_TEMP); s += vals[i]; }
      for (int off = 16; off > 0; off >>= 1) s += __shfl_xor(s, off, 32);
      const float inv = 1.0f / s;
      float*  orow = out + ((size_t)(bbase + row) * (L_ + 1) + t) * V_ + lane * 32;
      half_t* srow = sm + row * SB_STRIDE + lane * 32;
#pragma unroll
      for (int i = 0; i < 32; i++) {
        float p = vals[i] * inv;
        __builtin_nontemporal_store(p, orow + i);   // write-once -> keep L2 for weights
        srow[i] = (half_t)p;
      }
    }
    __syncthreads();

    // ---- e_new = sample @ W_emb + b_emb ------------------------------------
#pragma unroll
    for (int q = 0; q < 4; q++) {
      const int nb = (wave * 4 + q) * 16;
      v8f acc = {};
      for (int k0 = 0; k0 < V_; k0 += 32) {
        v16h a = load_frag(sm, SB_STRIDE, lane, k0);
        v16h b = load_frag(Wt_emb + (size_t)nb * V_, V_, lane, k0);
        acc = wmma16(a, b, acc);
      }
      const int n = nb + ncol;
      const float bias = b_emb[n];
#pragma unroll
      for (int i = 0; i < 8; i++) {
        int m = mrow + i;
        e_bf[m * EH_STRIDE + n] = (half_t)(acc[i] + bias);
      }
    }
    __syncthreads();
  }

  // ---- eos row: out[:, L, :] = one-hot(0) ----------------------------------
  for (int i = tid; i < TILE_B * V_; i += THREADS) {
    int r = i >> 10, c = i & 1023;
    float v = (c == 0) ? 1.0f : 0.0f;
    __builtin_nontemporal_store(
        v, out + ((size_t)(bbase + r) * (L_ + 1) + L_) * V_ + c);
  }
}

// ------------------------------- launcher ----------------------------------
extern "C" void kernel_launch(void* const* d_in, const int* in_sizes, int n_in,
                              void* d_out, int out_size, void* d_ws, size_t ws_size,
                              hipStream_t stream) {
  const float* x       = (const float*)d_in[0];
  const float* W_agent = (const float*)d_in[1];
  const float* b_agent = (const float*)d_in[2];
  const float* W_ih    = (const float*)d_in[3];
  const float* W_hh    = (const float*)d_in[4];
  const float* b_ih    = (const float*)d_in[5];
  const float* b_hh    = (const float*)d_in[6];
  const float* W_out   = (const float*)d_in[7];
  const float* b_out   = (const float*)d_in[8];
  const float* W_emb   = (const float*)d_in[9];
  const float* b_emb   = (const float*)d_in[10];
  const float* sos     = (const float*)d_in[11];
  const float* gumbel  = (const float*)d_in[12];

  half_t* ws = (half_t*)d_ws;
  half_t* Wt_agent = ws;                                  // [H][DIN]
  half_t* Wb_ih    = Wt_agent + (size_t)H_ * DIN;         // [3H][E]
  half_t* Wb_hh    = Wb_ih    + (size_t)H3 * E_;          // [3H][H]
  half_t* Wt_out   = Wb_hh    + (size_t)H3 * H_;          // [V][H]
  half_t* Wt_emb   = Wt_out   + (size_t)V_ * H_;          // [E][V]

  auto blocks = [](int n) { return (n + 255) / 256; };
  cvt_transpose_f16<<<blocks(DIN * H_), 256, 0, stream>>>(W_agent, Wt_agent, DIN, H_);
  cvt_copy_f16     <<<blocks(H3 * E_),  256, 0, stream>>>(W_ih, Wb_ih, H3 * E_);
  cvt_copy_f16     <<<blocks(H3 * H_),  256, 0, stream>>>(W_hh, Wb_hh, H3 * H_);
  cvt_transpose_f16<<<blocks(H_ * V_),  256, 0, stream>>>(W_out, Wt_out, H_, V_);
  cvt_transpose_f16<<<blocks(V_ * E_),  256, 0, stream>>>(W_emb, Wt_emb, V_, E_);

  const size_t shmem =
      (size_t)TILE_B * EH_STRIDE * sizeof(half_t) * 2 +   // e_bf, h_bf
      (size_t)TILE_B * LG_STRIDE * sizeof(float) +        // logits / gumbel stage
      (size_t)TILE_B * SB_STRIDE * sizeof(half_t);        // sample / x stage

  (void)hipFuncSetAttribute((const void*)rnn_sender_kernel,
                            hipFuncAttributeMaxDynamicSharedMemorySize,
                            (int)shmem);

  rnn_sender_kernel<<<B_ / TILE_B, THREADS, shmem, stream>>>(
      x, b_agent, b_ih, b_hh, b_out, b_emb, sos, gumbel,
      Wt_agent, Wb_ih, Wb_hh, Wt_out, Wt_emb, (float*)d_out);
}
